// Fractal_EGNN_v2_18279380812420
// MI455X (gfx1250) — matrix operand order repaired and checked by reading, not compiled, not run on hardware
//
#include <hip/hip_runtime.h>

#define NNODES  20000
#define NEDGES  160000
#define NB      16
#define DD      128
#define NLAYERS 20
#define LROW    136   // padded LDS row stride in bf16 elements (128 + 8)

typedef __attribute__((ext_vector_type(16))) __bf16 bf16x16;
typedef __attribute__((ext_vector_type(8)))  __bf16 bf16x8;
typedef __attribute__((ext_vector_type(8)))  float  f32x8;

// ---------------------------------------------------------------- helpers

__device__ __forceinline__ float silu_f(float y) {
  return y / (1.0f + __expf(-y));
}

__device__ __forceinline__ void fatomic_add(float* p, float v) {
  __hip_atomic_fetch_add(p, v, __ATOMIC_RELAXED, __HIP_MEMORY_SCOPE_AGENT);
}

// A fragment (16x32 bf16): per ISA, lane l (row M = l%16) needs two contiguous
// 8-element K runs: [kb + (l/16)*8, +8) and [kb + 16 + (l/16)*8, +8).
__device__ __forceinline__ bf16x16 load_afrag(const __bf16* rowp, int kb, int q) {
  bf16x8 a0 = *(const bf16x8*)(rowp + kb + q * 8);
  bf16x8 a1 = *(const bf16x8*)(rowp + kb + 16 + q * 8);
  return __builtin_shufflevector(a0, a1, 0,1,2,3,4,5,6,7,8,9,10,11,12,13,14,15);
}

// One K=32 step over all 8 N tiles. B fragments are prepacked so each lane
// loads 16 contiguous bf16 (32 bytes).
__device__ __forceinline__ void gemm_k32(f32x8 C[8], bf16x16 A,
                                         const __bf16* __restrict__ wp,
                                         int kt, int lane) {
#pragma unroll
  for (int nt = 0; nt < 8; ++nt) {
    bf16x16 B = *(const bf16x16*)(wp + (((kt << 3) + nt) << 9) + lane * 16);
    C[nt] = __builtin_amdgcn_wmma_f32_16x16x32_bf16(
        false, A, false, B, (short)0, C[nt], false, false);
  }
}

__device__ __forceinline__ void zero_acc(f32x8 C[8]) {
#pragma unroll
  for (int i = 0; i < 8; ++i)
#pragma unroll
    for (int j = 0; j < 8; ++j) C[i][j] = 0.0f;
}

// ---------------------------------------------------------------- prepack
// Pack f32 weight [K x 128] (row-major) into WMMA-B fragment order, bf16.
// frag (kt,nt): element (lane,j) = W[kt*32 + (j/8)*16 + (lane/16)*8 + j%8][nt*16 + lane%16]
__global__ void pack_kernel(const float* __restrict__ src, __bf16* __restrict__ dst,
                            int K, int srcStride, int total) {
  int tid = blockIdx.x * blockDim.x + threadIdx.x;
  if (tid >= total) return;
  int per = K * DD;
  int layer = tid / per;
  int e = tid - layer * per;
  int frag = e >> 9;
  int within = e & 511;
  int lane = within >> 4, j = within & 15;
  int kt = frag >> 3, nt = frag & 7;
  int k = kt * 32 + (j >> 3) * 16 + (lane >> 4) * 8 + (j & 7);
  int n = nt * 16 + (lane & 15);
  dst[(size_t)layer * per + e] = (__bf16)src[(size_t)layer * srcStride + k * DD + n];
}

// ---------------------------------------------------------------- embed
__global__ void embed_kernel(const float* __restrict__ x, const float* __restrict__ embW,
                             const float* __restrict__ embB, float* __restrict__ h,
                             float* __restrict__ h0, __bf16* __restrict__ hb, int total) {
  int tid = blockIdx.x * blockDim.x + threadIdx.x;
  if (tid >= total) return;
  int n = tid >> 7, d = tid & 127;
  float v = x[n] * embW[d] + embB[d];
  h[tid] = v;
  h0[tid] = v;
  hb[tid] = (__bf16)v;
}

__global__ void zero_kernel(float* __restrict__ p, int n) {
  int i = blockIdx.x * blockDim.x + threadIdx.x;
  if (i < n) p[i] = 0.0f;
}

// ---------------------------------------------------------------- message
// One block = 64 edges, 128 threads = 4 waves, each wave owns 16 edges x 128 feats.
__global__ __launch_bounds__(128) void msg_kernel(
    const __bf16* __restrict__ hb, const float* __restrict__ pos,
    const int* __restrict__ ei,
    const __bf16* __restrict__ w1p, const float* __restrict__ w1last,
    const float* __restrict__ b1v, const float* __restrict__ g1v, const float* __restrict__ s1v,
    const __bf16* __restrict__ w2p,
    const float* __restrict__ b2v, const float* __restrict__ g2v, const float* __restrict__ s2v,
    float* __restrict__ agg) {
  __shared__ __bf16 Adst[64 * LROW];
  __shared__ __bf16 Asrc[64 * LROW];
  __shared__ __bf16 Am[64 * LROW];
  __shared__ float distS[64];

  const int t = threadIdx.x;
  const int e0 = blockIdx.x * 64;

  // prefetch GEMM2 weights (32 KB) while we gather
  __builtin_prefetch(w2p + (size_t)t * 128, 0, 3);

  if (t < 64) {
    int e = e0 + t;
    int sN = ei[e];           // src
    int dN = ei[NEDGES + e];  // dst
    float dx = pos[dN * 3 + 0] - pos[sN * 3 + 0];
    float dy = pos[dN * 3 + 1] - pos[sN * 3 + 1];
    float dz = pos[dN * 3 + 2] - pos[sN * 3 + 2];
    distS[t] = sqrtf(dx * dx + dy * dy + dz * dz);
  }
  // gather 128 bf16 rows (64 dst + 64 src): 8 rows per iter, 16B chunk per thread
#pragma unroll 4
  for (int it = 0; it < 16; ++it) {
    int row = it * 8 + (t >> 4);
    int chunk = t & 15;
    int el = row & 63;
    int e = e0 + el;
    int node = (row < 64) ? ei[NEDGES + e] : ei[e];  // dst rows first, then src
    bf16x8 v = *(const bf16x8*)(hb + (size_t)node * DD + chunk * 8);
    *(bf16x8*)((row < 64 ? Adst : Asrc) + el * LROW + chunk * 8) = v;
  }
  __syncthreads();

  const int wv = t >> 5, lane = t & 31;
  const int q = lane >> 4, ln = lane & 15;
  const int mbase = wv * 16;
  const __bf16* rowD = Adst + (mbase + ln) * LROW;
  const __bf16* rowS = Asrc + (mbase + ln) * LROW;

  f32x8 C[8];
  zero_acc(C);

  // GEMM1: [h_dst | h_src] (K=256) x Wm1[0:256]
#pragma unroll
  for (int kt = 0; kt < 8; ++kt) {
    const __bf16* rowp = (kt < 4) ? rowD : rowS;
    bf16x16 A = load_afrag(rowp, (kt & 3) * 32, q);
    gemm_k32(C, A, w1p, kt, lane);
  }

  // epilogue 1: + dist * Wm1[256] + bias -> LN -> SiLU -> Am (bf16)
  {
    float wl[8], bb[8], gg[8], ss[8];
#pragma unroll
    for (int nt = 0; nt < 8; ++nt) {
      int n = nt * 16 + ln;
      wl[nt] = w1last[n]; bb[nt] = b1v[n]; gg[nt] = g1v[n]; ss[nt] = s1v[n];
    }
#pragma unroll
    for (int r = 0; r < 8; ++r) {
      int el = mbase + q * 8 + r;
      float dist = distS[el];
      float xv[8];
      float sum = 0.0f;
#pragma unroll
      for (int nt = 0; nt < 8; ++nt) {
        xv[nt] = C[nt][r] + dist * wl[nt] + bb[nt];
        sum += xv[nt];
      }
      sum += __shfl_xor(sum, 1, 32); sum += __shfl_xor(sum, 2, 32);
      sum += __shfl_xor(sum, 4, 32); sum += __shfl_xor(sum, 8, 32);
      float mean = sum * (1.0f / 128.0f);
      float vs = 0.0f;
#pragma unroll
      for (int nt = 0; nt < 8; ++nt) { float dv = xv[nt] - mean; vs += dv * dv; }
      vs += __shfl_xor(vs, 1, 32); vs += __shfl_xor(vs, 2, 32);
      vs += __shfl_xor(vs, 4, 32); vs += __shfl_xor(vs, 8, 32);
      float rstd = rsqrtf(vs * (1.0f / 128.0f) + 1e-5f);
#pragma unroll
      for (int nt = 0; nt < 8; ++nt) {
        float yv = (xv[nt] - mean) * rstd * gg[nt] + ss[nt];
        Am[el * LROW + nt * 16 + ln] = (__bf16)silu_f(yv);
      }
    }
  }
  __syncthreads();

  // GEMM2: m (K=128) x Wm2
  zero_acc(C);
  const __bf16* rowM = Am + (mbase + ln) * LROW;
#pragma unroll
  for (int kt = 0; kt < 4; ++kt) {
    bf16x16 A = load_afrag(rowM, kt * 32, q);
    gemm_k32(C, A, w2p, kt, lane);
  }

  // epilogue 2: + bias -> LN -> SiLU -> atomic scatter-add into agg[dst]
  {
    float bb[8], gg[8], ss[8];
#pragma unroll
    for (int nt = 0; nt < 8; ++nt) {
      int n = nt * 16 + ln;
      bb[nt] = b2v[n]; gg[nt] = g2v[n]; ss[nt] = s2v[n];
    }
#pragma unroll
    for (int r = 0; r < 8; ++r) {
      int el = mbase + q * 8 + r;
      float xv[8];
      float sum = 0.0f;
#pragma unroll
      for (int nt = 0; nt < 8; ++nt) { xv[nt] = C[nt][r] + bb[nt]; sum += xv[nt]; }
      sum += __shfl_xor(sum, 1, 32); sum += __shfl_xor(sum, 2, 32);
      sum += __shfl_xor(sum, 4, 32); sum += __shfl_xor(sum, 8, 32);
      float mean = sum * (1.0f / 128.0f);
      float vs = 0.0f;
#pragma unroll
      for (int nt = 0; nt < 8; ++nt) { float dv = xv[nt] - mean; vs += dv * dv; }
      vs += __shfl_xor(vs, 1, 32); vs += __shfl_xor(vs, 2, 32);
      vs += __shfl_xor(vs, 4, 32); vs += __shfl_xor(vs, 8, 32);
      float rstd = rsqrtf(vs * (1.0f / 128.0f) + 1e-5f);
      int dnode = ei[NEDGES + e0 + el];
#pragma unroll
      for (int nt = 0; nt < 8; ++nt) {
        float yv = (xv[nt] - mean) * rstd * gg[nt] + ss[nt];
        fatomic_add(&agg[(size_t)dnode * DD + nt * 16 + ln], silu_f(yv));
      }
    }
  }
}

// ---------------------------------------------------------------- update
// One block = 64 nodes. u = [h | agg] -> MLP -> (optional +h0) -> h, hb, h0.
__global__ __launch_bounds__(128) void upd_kernel(
    float* __restrict__ hmem, const float* __restrict__ agg,
    const __bf16* __restrict__ w1p,
    const float* __restrict__ b1v, const float* __restrict__ g1v, const float* __restrict__ s1v,
    const __bf16* __restrict__ w2p,
    const float* __restrict__ b2v, const float* __restrict__ g2v, const float* __restrict__ s2v,
    float* __restrict__ h0, __bf16* __restrict__ hb, int addRes) {
  __shared__ __bf16 Ah[64 * LROW];
  __shared__ __bf16 Ag[64 * LROW];
  __shared__ __bf16 Am[64 * LROW];

  const int t = threadIdx.x;
  const int n0 = blockIdx.x * 64;

  // gather: rows 0-63 = h (f32 -> bf16), rows 64-127 = agg
#pragma unroll 4
  for (int it = 0; it < 16; ++it) {
    int row = it * 8 + (t >> 4);
    int chunk = t & 15;
    int nloc = row & 63;
    int node = n0 + nloc;
    if (node > NNODES - 1) node = NNODES - 1;
    const float* sp = (row < 64 ? hmem : agg) + (size_t)node * DD + chunk * 8;
    bf16x8 o;
#pragma unroll
    for (int j = 0; j < 8; ++j) o[j] = (__bf16)sp[j];
    *(bf16x8*)((row < 64 ? Ah : Ag) + nloc * LROW + chunk * 8) = o;
  }
  __syncthreads();

  const int wv = t >> 5, lane = t & 31;
  const int q = lane >> 4, ln = lane & 15;
  const int mbase = wv * 16;
  const __bf16* rowH = Ah + (mbase + ln) * LROW;
  const __bf16* rowG = Ag + (mbase + ln) * LROW;

  f32x8 C[8];
  zero_acc(C);

  // GEMM1: [h | agg] (K=256) x Wu1
#pragma unroll
  for (int kt = 0; kt < 8; ++kt) {
    const __bf16* rowp = (kt < 4) ? rowH : rowG;
    bf16x16 A = load_afrag(rowp, (kt & 3) * 32, q);
    gemm_k32(C, A, w1p, kt, lane);
  }

  // epilogue 1: bias -> LN -> SiLU -> Am
  {
    float bb[8], gg[8], ss[8];
#pragma unroll
    for (int nt = 0; nt < 8; ++nt) {
      int n = nt * 16 + ln;
      bb[nt] = b1v[n]; gg[nt] = g1v[n]; ss[nt] = s1v[n];
    }
#pragma unroll
    for (int r = 0; r < 8; ++r) {
      int el = mbase + q * 8 + r;
      float xv[8];
      float sum = 0.0f;
#pragma unroll
      for (int nt = 0; nt < 8; ++nt) { xv[nt] = C[nt][r] + bb[nt]; sum += xv[nt]; }
      sum += __shfl_xor(sum, 1, 32); sum += __shfl_xor(sum, 2, 32);
      sum += __shfl_xor(sum, 4, 32); sum += __shfl_xor(sum, 8, 32);
      float mean = sum * (1.0f / 128.0f);
      float vs = 0.0f;
#pragma unroll
      for (int nt = 0; nt < 8; ++nt) { float dv = xv[nt] - mean; vs += dv * dv; }
      vs += __shfl_xor(vs, 1, 32); vs += __shfl_xor(vs, 2, 32);
      vs += __shfl_xor(vs, 4, 32); vs += __shfl_xor(vs, 8, 32);
      float rstd = rsqrtf(vs * (1.0f / 128.0f) + 1e-5f);
#pragma unroll
      for (int nt = 0; nt < 8; ++nt) {
        float yv = (xv[nt] - mean) * rstd * gg[nt] + ss[nt];
        Am[el * LROW + nt * 16 + ln] = (__bf16)silu_f(yv);
      }
    }
  }
  __syncthreads();

  // GEMM2: (K=128) x Wu2
  zero_acc(C);
  const __bf16* rowM = Am + (mbase + ln) * LROW;
#pragma unroll
  for (int kt = 0; kt < 4; ++kt) {
    bf16x16 A = load_afrag(rowM, kt * 32, q);
    gemm_k32(C, A, w2p, kt, lane);
  }

  // epilogue 2: bias -> LN -> SiLU -> (+h0) -> h, hb (and refresh h0)
  {
    float bb[8], gg[8], ss[8];
#pragma unroll
    for (int nt = 0; nt < 8; ++nt) {
      int n = nt * 16 + ln;
      bb[nt] = b2v[n]; gg[nt] = g2v[n]; ss[nt] = s2v[n];
    }
#pragma unroll
    for (int r = 0; r < 8; ++r) {
      int nloc = mbase + q * 8 + r;
      int node = n0 + nloc;
      float xv[8];
      float sum = 0.0f;
#pragma unroll
      for (int nt = 0; nt < 8; ++nt) { xv[nt] = C[nt][r] + bb[nt]; sum += xv[nt]; }
      sum += __shfl_xor(sum, 1, 32); sum += __shfl_xor(sum, 2, 32);
      sum += __shfl_xor(sum, 4, 32); sum += __shfl_xor(sum, 8, 32);
      float mean = sum * (1.0f / 128.0f);
      float vs = 0.0f;
#pragma unroll
      for (int nt = 0; nt < 8; ++nt) { float dv = xv[nt] - mean; vs += dv * dv; }
      vs += __shfl_xor(vs, 1, 32); vs += __shfl_xor(vs, 2, 32);
      vs += __shfl_xor(vs, 4, 32); vs += __shfl_xor(vs, 8, 32);
      float rstd = rsqrtf(vs * (1.0f / 128.0f) + 1e-5f);
      if (node < NNODES) {
#pragma unroll
        for (int nt = 0; nt < 8; ++nt) {
          float yv = (xv[nt] - mean) * rstd * gg[nt] + ss[nt];
          float o = silu_f(yv);
          size_t idx = (size_t)node * DD + nt * 16 + ln;
          if (addRes) o += h0[idx];
          hmem[idx] = o;
          hb[idx] = (__bf16)o;
          if (addRes) h0[idx] = o;
        }
      }
    }
  }
}

// ---------------------------------------------------------------- pool + head
__global__ void pool_kernel(const float* __restrict__ h, const int* __restrict__ batch,
                            float* __restrict__ pooled, int total) {
  int tid = blockIdx.x * blockDim.x + threadIdx.x;
  if (tid >= total) return;
  int n = tid >> 7, d = tid & 127;
  fatomic_add(&pooled[batch[n] * DD + d], h[tid]);
}

__global__ void head_kernel(const float* __restrict__ pooled,
                            const float* __restrict__ W1, const float* __restrict__ b1,
                            const float* __restrict__ W2, const float* __restrict__ b2,
                            float* __restrict__ out) {
  __shared__ float red[DD];
  int t = threadIdx.x;
  for (int b = 0; b < NB; ++b) {
    float acc = b1[t];
    for (int k = 0; k < DD; ++k) acc += pooled[b * DD + k] * W1[k * DD + t];
    acc = fmaxf(acc, 0.0f);
    red[t] = acc * W2[t];
    __syncthreads();
    for (int s = 64; s > 0; s >>= 1) {
      if (t < s) red[t] += red[t + s];
      __syncthreads();
    }
    if (t == 0) out[b] = red[0] + b2[0];
    __syncthreads();
  }
}

// ---------------------------------------------------------------- launch

extern "C" void kernel_launch(void* const* d_in, const int* in_sizes, int n_in,
                              void* d_out, int out_size, void* d_ws, size_t ws_size,
                              hipStream_t stream) {
  const float* x    = (const float*)d_in[0];
  const float* pos  = (const float*)d_in[1];
  const int* eis[4] = {(const int*)d_in[2], (const int*)d_in[3],
                       (const int*)d_in[4], (const int*)d_in[5]};
  const int* batch  = (const int*)d_in[6];
  const float* embW = (const float*)d_in[7];
  const float* embB = (const float*)d_in[8];
  const float* Wm1  = (const float*)d_in[9];
  const float* bm1  = (const float*)d_in[10];
  const float* gm1  = (const float*)d_in[11];
  const float* sm1  = (const float*)d_in[12];
  const float* Wm2  = (const float*)d_in[13];
  const float* bm2  = (const float*)d_in[14];
  const float* gm2  = (const float*)d_in[15];
  const float* sm2  = (const float*)d_in[16];
  const float* Wu1  = (const float*)d_in[17];
  const float* bu1  = (const float*)d_in[18];
  const float* gu1  = (const float*)d_in[19];
  const float* su1  = (const float*)d_in[20];
  const float* Wu2  = (const float*)d_in[21];
  const float* bu2  = (const float*)d_in[22];
  const float* gu2  = (const float*)d_in[23];
  const float* su2  = (const float*)d_in[24];
  const float* pW1  = (const float*)d_in[25];
  const float* pb1  = (const float*)d_in[26];
  const float* pW2  = (const float*)d_in[27];
  const float* pb2  = (const float*)d_in[28];

  char* wsp = (char*)d_ws;
  auto carve = [&](size_t bytes) -> char* {
    char* p = wsp;
    wsp += (bytes + 255) & ~(size_t)255;
    return p;
  };
  float*  h      = (float*)carve((size_t)NNODES * DD * 4);
  float*  h0     = (float*)carve((size_t)NNODES * DD * 4);
  float*  agg    = (float*)carve((size_t)NNODES * DD * 4);
  float*  pooled = (float*)carve((size_t)NB * DD * 4);
  __bf16* hb     = (__bf16*)carve((size_t)NNODES * DD * 2);
  __bf16* wm1p   = (__bf16*)carve((size_t)NLAYERS * 256 * DD * 2);
  __bf16* wm2p   = (__bf16*)carve((size_t)NLAYERS * 128 * DD * 2);
  __bf16* wu1p   = (__bf16*)carve((size_t)NLAYERS * 256 * DD * 2);
  __bf16* wu2p   = (__bf16*)carve((size_t)NLAYERS * 128 * DD * 2);

  // prepack weights (bf16, WMMA B-fragment order)
  {
    int t1 = NLAYERS * 256 * DD;
    int t2 = NLAYERS * 128 * DD;
    pack_kernel<<<(t1 + 255) / 256, 256, 0, stream>>>(Wm1, wm1p, 256, 257 * DD, t1);
    pack_kernel<<<(t2 + 255) / 256, 256, 0, stream>>>(Wm2, wm2p, 128, 128 * DD, t2);
    pack_kernel<<<(t1 + 255) / 256, 256, 0, stream>>>(Wu1, wu1p, 256, 256 * DD, t1);
    pack_kernel<<<(t2 + 255) / 256, 256, 0, stream>>>(Wu2, wu2p, 128, 128 * DD, t2);
  }

  int totN = NNODES * DD;
  embed_kernel<<<(totN + 255) / 256, 256, 0, stream>>>(x, embW, embB, h, h0, hb, totN);

  for (int l = 0; l < NLAYERS; ++l) {
    int r = l & 3;
    zero_kernel<<<(totN + 255) / 256, 256, 0, stream>>>(agg, totN);
    msg_kernel<<<NEDGES / 64, 128, 0, stream>>>(
        hb, pos, eis[r],
        wm1p + (size_t)l * 256 * DD,
        Wm1 + (size_t)l * 257 * DD + 256 * DD,   // dist row of Wm1
        bm1 + l * DD, gm1 + l * DD, sm1 + l * DD,
        wm2p + (size_t)l * 128 * DD,
        bm2 + l * DD, gm2 + l * DD, sm2 + l * DD,
        agg);
    upd_kernel<<<(NNODES + 63) / 64, 128, 0, stream>>>(
        h, agg,
        wu1p + (size_t)l * 256 * DD,
        bu1 + l * DD, gu1 + l * DD, su1 + l * DD,
        wu2p + (size_t)l * 128 * DD,
        bu2 + l * DD, gu2 + l * DD, su2 + l * DD,
        h0, hb, (r == 3) ? 1 : 0);
  }

  int totB = NB * DD;
  zero_kernel<<<(totB + 255) / 256, 256, 0, stream>>>(pooled, totB);
  pool_kernel<<<(totN + 255) / 256, 256, 0, stream>>>(h, batch, pooled, totN);
  head_kernel<<<1, DD, 0, stream>>>(pooled, pW1, pb1, pW2, pb2, (float*)d_out);
}
